// MoE_5952824673138
// MI455X (gfx1250) — matrix operand order repaired and checked by reading, not compiled, not run on hardware
//
#include <hip/hip_runtime.h>
#include <hip/hip_bf16.h>
#include <math.h>

// ---------------------------------------------------------------------------
// MoE top-2 (B=4, N=2048, DIM=1024, HIDDEN=4096, E=16, CAP=256) for gfx1250.
// Expert FFN on v_wmma_f32_16x16x32_bf16. Each 256-thread block (8 waves)
// computes a 64x256 C-tile; each wave owns 32x64 (8 accumulators -> 8 WMMA
// per 32-deep K-step). Global->register->LDS staging is software-pipelined
// so VMEM latency hides behind the WMMAs. fp32 weights are converted to bf16
// on the fly with packed converts and 128-bit LDS traffic.
// ---------------------------------------------------------------------------

typedef __attribute__((ext_vector_type(16))) __bf16 v16bf;
typedef __attribute__((ext_vector_type(8)))  float  v8f;

#define MB   4
#define MN   2048
#define MDIM 1024
#define MHID 4096
#define MNE  16
#define MCAP 256
#define NTOK (MB * MN)
#define NROW (MB * MCAP)   // 1024 rows per expert

#define LDA_PAD 40         // As row length (32 + 8 pad), keeps 16B alignment
#define LDB_PAD 264        // Bs row length (256 + 8 pad), keeps 16B alignment

static __device__ __forceinline__ __bf16 f2bf(float f) { return (__bf16)f; }

// A fragment (16x32 bf16, ISA 7.12.2): lane r=L&15, h=L>>4;
// halfwords 0..7 -> K=8h+j ; halfwords 8..15 -> K=16+8h+j.
static __device__ __forceinline__ v16bf frag_a(const __bf16* As_row, int h) {
    v16bf a;
#pragma unroll
    for (int j = 0; j < 8; ++j) a[j] = As_row[8 * h + j];
#pragma unroll
    for (int j = 0; j < 8; ++j) a[8 + j] = As_row[16 + 8 * h + j];
    return a;
}

// B fragment (32x16 bf16): lane column n=L&15, h=L>>4;
// halfwords 0..7 -> K=8h+j ; halfwords 8..15 -> K=16+8h+j.
static __device__ __forceinline__ v16bf frag_b(const __bf16* Bs_col, int h) {
    v16bf b;
#pragma unroll
    for (int j = 0; j < 8; ++j) b[j] = Bs_col[(8 * h + j) * LDB_PAD];
#pragma unroll
    for (int j = 0; j < 8; ++j) b[8 + j] = Bs_col[(16 + 8 * h + j) * LDB_PAD];
    return b;
}

// ---------------------------------------------------------------- zero scratch
__global__ void moe_zero_kernel(unsigned int* __restrict__ p, size_t nwords) {
    size_t i = (size_t)blockIdx.x * blockDim.x + threadIdx.x;
    size_t stride = (size_t)gridDim.x * blockDim.x;
    for (; i < nwords; i += stride) p[i] = 0u;
}

// ---------------------------------------------------------------- gating
__global__ void moe_gating_kernel(const float* __restrict__ x,
                                  const float* __restrict__ wg,
                                  float* __restrict__ tok_p,
                                  int*   __restrict__ tok_e1,
                                  int*   __restrict__ tok_e2,
                                  float* __restrict__ tok_g1,
                                  float* __restrict__ tok_g2) {
    __shared__ float xs[MDIM];
    __shared__ float lg[MNE];
    const int tok = blockIdx.x;
    const float* xr = x + (size_t)tok * MDIM;
    for (int d = threadIdx.x; d < MDIM; d += blockDim.x) xs[d] = xr[d];
    __syncthreads();
    if (threadIdx.x < MNE) {
        const int e = threadIdx.x;
        float acc = 0.f;
        for (int d = 0; d < MDIM; ++d) acc = fmaf(xs[d], wg[(size_t)d * MNE + e], acc);
        lg[e] = acc;
    }
    __syncthreads();
    if (threadIdx.x == 0) {
        float mx = lg[0];
        for (int e = 1; e < MNE; ++e) mx = fmaxf(mx, lg[e]);
        float p[MNE]; float s = 0.f;
        for (int e = 0; e < MNE; ++e) { p[e] = __expf(lg[e] - mx); s += p[e]; }
        const float inv = 1.f / s;
        int e1 = 0; float g1 = -1.f;
        for (int e = 0; e < MNE; ++e) { p[e] *= inv; if (p[e] > g1) { g1 = p[e]; e1 = e; } }
        int e2 = 0; float g2 = -1.f;
        for (int e = 0; e < MNE; ++e) if (e != e1 && p[e] > g2) { g2 = p[e]; e2 = e; }
        const float denom = g1 + g2 + 1e-9f;
        tok_e1[tok] = e1; tok_e2[tok] = e2;
        tok_g1[tok] = g1 / denom; tok_g2[tok] = g2 / denom;
        for (int e = 0; e < MNE; ++e) tok_p[(size_t)tok * MNE + e] = p[e];
    }
}

// ---------------------------------------------------------------- capacity scans
__global__ void moe_scan1_kernel(const int* __restrict__ tok_e1,
                                 float* __restrict__ tok_g1,
                                 int*   __restrict__ tok_p1,
                                 int*   __restrict__ cnt1,
                                 float* __restrict__ dens_cnt) {
    const int be = blockIdx.x, b = be >> 4, e = be & 15;
    const int lane = threadIdx.x;
    int run = 0;
    for (int n0 = 0; n0 < MN; n0 += 32) {
        const int tok = b * MN + n0 + lane;
        const bool m = (tok_e1[tok] == e);
        const unsigned mask = (unsigned)__ballot(m);
        if (m) {
            const int pos = run + __popc(mask & ((1u << lane) - 1u));
            if (pos < MCAP) tok_p1[tok] = pos;
            else { tok_p1[tok] = 0; tok_g1[tok] = 0.f; }
        }
        run += __popc(mask);
    }
    if (lane == 0) {
        cnt1[be] = run < MCAP ? run : MCAP;
        dens_cnt[be] = (float)run;
    }
}

__global__ void moe_scan2_kernel(const int* __restrict__ tok_e2,
                                 float* __restrict__ tok_g2,
                                 int*   __restrict__ tok_p2,
                                 const int* __restrict__ cnt1) {
    const int be = blockIdx.x, b = be >> 4, e = be & 15;
    const int lane = threadIdx.x;
    int run = cnt1[be];
    for (int n0 = 0; n0 < MN; n0 += 32) {
        const int tok = b * MN + n0 + lane;
        const bool m = (tok_e2[tok] == e);
        const unsigned mask = (unsigned)__ballot(m);
        if (m) {
            const int pos = run + __popc(mask & ((1u << lane) - 1u));
            if (pos < MCAP) tok_p2[tok] = pos;
            else { tok_p2[tok] = 0; tok_g2[tok] = 0.f; }
        }
        run += __popc(mask);
    }
}

// ---------------------------------------------------------------- dispatch
__global__ void moe_dispatch_kernel(const float* __restrict__ x,
                                    const int* __restrict__ tok_e1,
                                    const int* __restrict__ tok_e2,
                                    const float* __restrict__ tok_g1,
                                    const float* __restrict__ tok_g2,
                                    const int* __restrict__ tok_p1,
                                    const int* __restrict__ tok_p2,
                                    __bf16* __restrict__ Xe) {
    const int tok = blockIdx.x;
    const int b = tok / MN;
    const float g1 = tok_g1[tok], g2 = tok_g2[tok];
    __bf16* d1 = (g1 != 0.f)
        ? Xe + (((size_t)tok_e1[tok] * MB + b) * MCAP + tok_p1[tok]) * MDIM : nullptr;
    __bf16* d2 = (g2 != 0.f)
        ? Xe + (((size_t)tok_e2[tok] * MB + b) * MCAP + tok_p2[tok]) * MDIM : nullptr;
    const int d = threadIdx.x * 4;                       // 256 thr * 4 = 1024
    const float4 v = *(const float4*)(x + (size_t)tok * MDIM + d);
    __bf16 t[4];
    t[0] = f2bf(v.x); t[1] = f2bf(v.y); t[2] = f2bf(v.z); t[3] = f2bf(v.w);
    if (d1) *(unsigned long long*)(d1 + d) = *(unsigned long long*)t;
    if (d2) *(unsigned long long*)(d2 + d) = *(unsigned long long*)t;
}

// ---------------------------------------------------------------- WMMA FFN
// K      : reduction depth (and A row stride)
// NLD    : output row length (and B row stride)
// GELU   : apply exact GELU epilogue
// OutT   : __bf16 (hidden) or float (expert outputs)
template <int K, int NLD, bool GELU, typename OutT>
__global__ __launch_bounds__(256)
void moe_ffn_kernel(const __bf16* __restrict__ Abase,
                    const float* __restrict__ Wbase,
                    const float* __restrict__ bias,
                    OutT* __restrict__ Out) {
    __shared__ __bf16 As[64][LDA_PAD];
    __shared__ __bf16 Bs[32][LDB_PAD];
    const int e = blockIdx.z;
    const int row0 = blockIdx.y * 64;
    const int col0 = blockIdx.x * 256;
    const __bf16* A  = Abase + (size_t)e * NROW * K;
    const float*  Bw = Wbase + (size_t)e * K * NLD;
    const int tid = threadIdx.x, lane = tid & 31, wave = tid >> 5;
    const int wm = wave >> 2, wn = wave & 3;
    const int r = lane & 15, h = lane >> 4;
    const int a_rr = tid >> 2, a_cc = (tid & 3) * 8;     // 64x32, 8 bf16/thread
    const int b_rr = tid >> 3, b_cc = (tid & 7) * 32;    // 32x256, 32 f32/thread

    v8f acc[2][4] = {};

    const __bf16* aPtr = &A[(size_t)(row0 + a_rr) * K + a_cc];
    const float*  bPtr = &Bw[(size_t)b_rr * NLD + col0 + b_cc];

    // prologue: stage K-step 0 into registers
    uint4  aReg = *(const uint4*)aPtr;
    float4 bReg[8];
#pragma unroll
    for (int i = 0; i < 8; ++i) bReg[i] = ((const float4*)bPtr)[i];

    for (int kt = 0; kt < K; kt += 32) {
        // commit staged registers to LDS (fp32 -> bf16 packed converts)
        *(uint4*)&As[a_rr][a_cc] = aReg;
        {
            __bf16 t[32];
#pragma unroll
            for (int i = 0; i < 8; ++i) {
                t[4 * i + 0] = f2bf(bReg[i].x); t[4 * i + 1] = f2bf(bReg[i].y);
                t[4 * i + 2] = f2bf(bReg[i].z); t[4 * i + 3] = f2bf(bReg[i].w);
            }
#pragma unroll
            for (int i = 0; i < 4; ++i)
                *(uint4*)&Bs[b_rr][b_cc + 8 * i] = *(uint4*)&t[8 * i];
        }
        __syncthreads();
        // issue next K-step's global loads now; they retire behind the WMMAs
        if (kt + 32 < K) {
            aReg = *(const uint4*)(aPtr + kt + 32);
#pragma unroll
            for (int i = 0; i < 8; ++i)
                bReg[i] = ((const float4*)(bPtr + (size_t)(kt + 32) * NLD))[i];
        }
        if (kt + 64 < K) {
            __builtin_prefetch(bPtr + (size_t)(kt + 64) * NLD, 0, 1);
            __builtin_prefetch(aPtr + kt + 64, 0, 1);
        }
        // fragments + 8 WMMAs (wave tile 32x64)
        const v16bf a0 = frag_a(&As[wm * 32 + r][0], h);
        const v16bf a1 = frag_a(&As[wm * 32 + 16 + r][0], h);
        v16bf bf[4];
#pragma unroll
        for (int nf = 0; nf < 4; ++nf) bf[nf] = frag_b(&Bs[0][wn * 64 + nf * 16 + r], h);
#pragma unroll
        for (int nf = 0; nf < 4; ++nf)
            acc[0][nf] = __builtin_amdgcn_wmma_f32_16x16x32_bf16(
                false, a0, false, bf[nf], (short)0, acc[0][nf], false, false);
#pragma unroll
        for (int nf = 0; nf < 4; ++nf)
            acc[1][nf] = __builtin_amdgcn_wmma_f32_16x16x32_bf16(
                false, a1, false, bf[nf], (short)0, acc[1][nf], false, false);
        __syncthreads();
    }
    // epilogue: bias (+ exact GELU), store
#pragma unroll
    for (int mi = 0; mi < 2; ++mi)
#pragma unroll
        for (int nf = 0; nf < 4; ++nf)
#pragma unroll
            for (int i = 0; i < 8; ++i) {
                const int M = row0 + wm * 32 + mi * 16 + h * 8 + i;
                const int Nc = col0 + wn * 64 + nf * 16 + r;
                float v = acc[mi][nf][i] + bias[Nc];
                if (GELU) v = 0.5f * v * (1.0f + erff(v * 0.70710678118654752f));
                Out[((size_t)e * NROW + M) * NLD + Nc] = (OutT)v;
            }
}

// ---------------------------------------------------------------- combine
__global__ void moe_combine_kernel(const float* __restrict__ Eout,
                                   const int* __restrict__ tok_e1,
                                   const int* __restrict__ tok_e2,
                                   const float* __restrict__ tok_g1,
                                   const float* __restrict__ tok_g2,
                                   const int* __restrict__ tok_p1,
                                   const int* __restrict__ tok_p2,
                                   float* __restrict__ out) {
    const int tok = blockIdx.x;
    const int b = tok / MN;
    const float g1 = tok_g1[tok], g2 = tok_g2[tok];
    const float* s1 = Eout + (((size_t)tok_e1[tok] * MB + b) * MCAP + tok_p1[tok]) * MDIM;
    const float* s2 = Eout + (((size_t)tok_e2[tok] * MB + b) * MCAP + tok_p2[tok]) * MDIM;
    const int d = threadIdx.x * 4;
    float4 v = make_float4(0.f, 0.f, 0.f, 0.f);
    if (g1 != 0.f) {
        const float4 a = *(const float4*)(s1 + d);
        v.x += g1 * a.x; v.y += g1 * a.y; v.z += g1 * a.z; v.w += g1 * a.w;
    }
    if (g2 != 0.f) {
        const float4 a = *(const float4*)(s2 + d);
        v.x += g2 * a.x; v.y += g2 * a.y; v.z += g2 * a.z; v.w += g2 * a.w;
    }
    *(float4*)(out + (size_t)tok * MDIM + d) = v;
}

// ---------------------------------------------------------------- aux loss
__global__ void moe_loss_kernel(const float* __restrict__ tok_p,
                                const float* __restrict__ dens_cnt,
                                float* __restrict__ out_loss) {
    __shared__ float part[64];
    const int t = threadIdx.x;
    const int b = t >> 4, e = t & 15;
    float s = 0.f;
    for (int n = 0; n < MN; ++n) s += tok_p[((size_t)(b * MN + n)) * MNE + e];
    part[t] = (s / (float)MN) * (dens_cnt[t] / (float)MN);
    __syncthreads();
    if (t == 0) {
        float tot = 0.f;
        for (int i = 0; i < 64; ++i) tot += part[i];
        out_loss[0] = (tot / 64.0f) * (float)(MNE * MNE) * 0.01f;
    }
}

// ---------------------------------------------------------------- launcher
extern "C" void kernel_launch(void* const* d_in, const int* in_sizes, int n_in,
                              void* d_out, int out_size, void* d_ws, size_t ws_size,
                              hipStream_t stream) {
    (void)in_sizes; (void)n_in; (void)out_size; (void)ws_size;
    const float* x  = (const float*)d_in[0];
    const float* wg = (const float*)d_in[1];
    const float* w1 = (const float*)d_in[2];
    const float* w2 = (const float*)d_in[3];
    const float* b1 = (const float*)d_in[4];
    const float* b2 = (const float*)d_in[5];
    float* out = (float*)d_out;

    char* w = (char*)d_ws;
    size_t off = 0;
    auto take = [&](size_t bytes) {
        size_t o = off; off += (bytes + 255) & ~(size_t)255; return o;
    };
    const size_t o_tokp = take((size_t)NTOK * MNE * 4);
    const size_t o_e1   = take((size_t)NTOK * 4);
    const size_t o_e2   = take((size_t)NTOK * 4);
    const size_t o_g1   = take((size_t)NTOK * 4);
    const size_t o_g2   = take((size_t)NTOK * 4);
    const size_t o_p1   = take((size_t)NTOK * 4);
    const size_t o_p2   = take((size_t)NTOK * 4);
    const size_t o_cnt1 = take(64 * 4);
    const size_t o_dens = take(64 * 4);
    const size_t xe_bytes = (size_t)MNE * MB * MCAP * MDIM * 2;     // 32 MB bf16
    const size_t o_Xe   = take(xe_bytes);
    const size_t o_H    = take((size_t)MNE * MB * MCAP * MHID * 2); // 128 MB bf16
    const size_t o_Eo   = take((size_t)MNE * MB * MCAP * MDIM * 4); // 64 MB f32

    float* tok_p = (float*)(w + o_tokp);
    int* tok_e1 = (int*)(w + o_e1); int* tok_e2 = (int*)(w + o_e2);
    float* tok_g1 = (float*)(w + o_g1); float* tok_g2 = (float*)(w + o_g2);
    int* tok_p1 = (int*)(w + o_p1); int* tok_p2 = (int*)(w + o_p2);
    int* cnt1 = (int*)(w + o_cnt1); float* dens = (float*)(w + o_dens);
    __bf16* Xe = (__bf16*)(w + o_Xe);
    __bf16* H  = (__bf16*)(w + o_H);
    float* Eout = (float*)(w + o_Eo);

    moe_zero_kernel<<<2048, 256, 0, stream>>>((unsigned int*)Xe, xe_bytes / 4);
    moe_gating_kernel<<<NTOK, 256, 0, stream>>>(x, wg, tok_p, tok_e1, tok_e2,
                                                tok_g1, tok_g2);
    moe_scan1_kernel<<<64, 32, 0, stream>>>(tok_e1, tok_g1, tok_p1, cnt1, dens);
    moe_scan2_kernel<<<64, 32, 0, stream>>>(tok_e2, tok_g2, tok_p2, cnt1);
    moe_dispatch_kernel<<<NTOK, 256, 0, stream>>>(x, tok_e1, tok_e2, tok_g1,
                                                  tok_g2, tok_p1, tok_p2, Xe);
    moe_ffn_kernel<MDIM, MHID, true, __bf16>
        <<<dim3(MHID / 256, NROW / 64, MNE), 256, 0, stream>>>(Xe, w1, b1, H);
    moe_ffn_kernel<MHID, MDIM, false, float>
        <<<dim3(MDIM / 256, NROW / 64, MNE), 256, 0, stream>>>(H, w2, b2, Eout);
    moe_combine_kernel<<<NTOK, 256, 0, stream>>>(Eout, tok_e1, tok_e2, tok_g1,
                                                 tok_g2, tok_p1, tok_p2, out);
    moe_loss_kernel<<<1, 64, 0, stream>>>(tok_p, dens,
                                          out + (size_t)MB * MN * MDIM);
}
